// PatientOutcomeModelEmbedding_33363305955554
// MI455X (gfx1250) — matrix operand order repaired and checked by reading, not compiled, not run on hardware
//
#include <hip/hip_runtime.h>
#include <hip/hip_bf16.h>

typedef __attribute__((ext_vector_type(16))) _Float16 v16h;
typedef __attribute__((ext_vector_type(8)))  float    v8f;

#define N_NODES 100000
#define N_EDGES 1600000
#define BB 256
#define TT 128
#define HH 128

// ---------------------------------------------------------------------------
// Weight prep: out[n*K + k] = (f16) op(B)(k,n), op = transpose/offset variant.
// Makes all GEMM B-fragment loads contiguous f16 b128s.
// ---------------------------------------------------------------------------
__global__ void k_prep_B(const float* __restrict__ B, int ldb, int bTrans, int koff,
                         _Float16* __restrict__ out, int N, int K) {
  int i = blockIdx.x * blockDim.x + threadIdx.x;
  if (i >= N * K) return;
  int n = i / K, k = i - n * K;
  float v = bTrans ? B[(size_t)n * ldb + koff + k] : B[(size_t)(koff + k) * ldb + n];
  out[i] = (_Float16)v;
}

// ---------------------------------------------------------------------------
// WMMA f16 GEMM: C[M,N] = A[M,K](f32) @ Bh[N,K](f16,K-major) (+bias)(+addRows)(relu)
// block (32,2): 2 waves; each wave owns a 16x64 tile (4 accumulators, one A
// fragment feeds 4 v_wmma). grid = (M/32, N/64). K multiple of 32.
// addRows row index = (m >> rowShift) (power-of-two divisor, no int division).
// ---------------------------------------------------------------------------
__global__ void k_gemm_f16(const float* __restrict__ A, int lda,
                           const _Float16* __restrict__ Bh,
                           const float* __restrict__ bias,
                           const float* __restrict__ addRows, int rowShift, int addLd,
                           int relu,
                           float* __restrict__ C, int ldc,
                           int M, int N, int K)
{
  const int lane = threadIdx.x;
  const int mt   = blockIdx.x * 2 + threadIdx.y;
  const int m0   = mt * 16;
  const int n00  = blockIdx.y * 64;
  const int half = lane >> 4;
  const int l15  = lane & 15;
  if (m0 >= M) return;

  v8f acc[4] = {};
  for (int kk = 0; kk < K; kk += 32) {
    v16h a;   // A 16x32 fragment, contiguous K order (row = l15, K-halves by lane-half)
    {
      const float* ar = A + (size_t)(m0 + l15) * lda + kk + half * 16;
#pragma unroll
      for (int j = 0; j < 8; ++j) a[j]     = (_Float16)ar[j];
#pragma unroll
      for (int j = 0; j < 8; ++j) a[8 + j] = (_Float16)ar[8 + j];
    }
#pragma unroll
    for (int u = 0; u < 4; ++u) {
      v16h b;   // B 32x16 fragment: col = l15, K = kk + half*16 + j (contiguous f16)
      const _Float16* bp = Bh + (size_t)(n00 + u * 16 + l15) * K + kk + half * 16;
#pragma unroll
      for (int j = 0; j < 16; ++j) b[j] = bp[j];
      acc[u] = __builtin_amdgcn_wmma_f32_16x16x32_f16(false, a, false, b,
                                                      (short)0, acc[u], false, false);
    }
  }
#pragma unroll
  for (int u = 0; u < 4; ++u) {
    const int n = n00 + u * 16 + l15;
    const float badd = bias ? bias[n] : 0.0f;     // hoisted: depends on column only
#pragma unroll
    for (int r = 0; r < 8; ++r) {
      const int m = m0 + r + half * 8;
      float v = acc[u][r] + badd;
      if (addRows) v += addRows[(size_t)(m >> rowShift) * addLd + n];
      if (relu)    v = fmaxf(v, 0.0f);
      C[(size_t)m * ldc + n] = v;
    }
  }
}

// ------------------------------ utilities ----------------------------------
__global__ void k_fill(float* __restrict__ p, float v, size_t n) {
  size_t i = (size_t)blockIdx.x * blockDim.x + threadIdx.x;
  if (i < n) p[i] = v;
}
__global__ void k_add_vec(float* __restrict__ o, const float* __restrict__ a,
                          const float* __restrict__ b, int n) {
  int i = blockIdx.x * blockDim.x + threadIdx.x;
  if (i < n) o[i] = a[i] + b[i];
}
__global__ void k_deg_acc(const int* __restrict__ dst, float* __restrict__ deg, int nE) {
  int e = blockIdx.x * blockDim.x + threadIdx.x;
  if (e < nE) atomicAdd(&deg[dst[e]], 1.0f);
}
__global__ void k_rsqrt_inplace(float* __restrict__ p, int n) {
  int i = blockIdx.x * blockDim.x + threadIdx.x;
  if (i < n) p[i] = rsqrtf(p[i]);
}

// ------------------------- GCN scatter (segment_sum) -----------------------
__global__ void k_agg_init(const float* __restrict__ h, const float* __restrict__ dinv,
                           const float* __restrict__ bias, float* __restrict__ agg) {
  size_t i = (size_t)blockIdx.x * blockDim.x + threadIdx.x;
  if (i >= (size_t)N_NODES * HH) return;
  int n = (int)(i >> 7), f = (int)(i & 127);
  float d = dinv[n];
  agg[i] = d * d * h[i] + bias[f];     // self-loop + PyG post-aggregation bias
}
__global__ void k_agg_edges(const float* __restrict__ h, const float* __restrict__ dinv,
                            const int* __restrict__ src, const int* __restrict__ dst,
                            float* __restrict__ agg, int nE) {
  int e = blockIdx.x * 8 + (threadIdx.x >> 5);
  if (e >= nE) return;
  int lane = threadIdx.x & 31;
  int s = src[e], d = dst[e];
  float c = dinv[s] * dinv[d];
  const float4 hv = *(const float4*)(h + (size_t)s * HH + lane * 4);
  float* ap = agg + (size_t)d * HH + lane * 4;
  __builtin_prefetch(ap, 1, 1);        // global_prefetch_b8 on destination line
  atomicAdd(ap + 0, c * hv.x);
  atomicAdd(ap + 1, c * hv.y);
  atomicAdd(ap + 2, c * hv.z);
  atomicAdd(ap + 3, c * hv.w);
}

// --------------------------- BatchNorm (train-mode) ------------------------
__global__ void k_bn_stats(const float* __restrict__ h, float* __restrict__ sums) {
  int f  = threadIdx.x;                // 0..127 (coalesced across features)
  int n0 = blockIdx.x * 256;
  int n1 = n0 + 256 < N_NODES ? n0 + 256 : N_NODES;
  float s = 0.f, ss = 0.f;
  for (int n = n0; n < n1; ++n) { float v = h[(size_t)n * HH + f]; s += v; ss += v * v; }
  atomicAdd(&sums[f], s);
  atomicAdd(&sums[HH + f], ss);
}
__global__ void k_bn_relu(const float* __restrict__ hin, const float* __restrict__ sums,
                          const float* __restrict__ gamma, const float* __restrict__ beta,
                          float* __restrict__ hout) {
  size_t i = (size_t)blockIdx.x * blockDim.x + threadIdx.x;
  if (i >= (size_t)N_NODES * HH) return;
  int f = (int)(i & 127);
  float mean = sums[f] / (float)N_NODES;
  float var  = sums[HH + f] / (float)N_NODES - mean * mean;
  float v = (hin[i] - mean) * rsqrtf(var + 1e-5f) * gamma[f] + beta[f];
  hout[i] = fmaxf(v, 0.0f);
}

// ----------------- exact-match id join (argmax of equality) ----------------
__global__ void k_join(const int* __restrict__ gpid, const int* __restrict__ pid,
                       const float* __restrict__ node_emb, float* __restrict__ fg) {
  __shared__ int best;
  if (threadIdx.x == 0) best = 0x7fffffff;
  __syncthreads();
  int b = blockIdx.x, tgt = pid[b];
  for (int n = threadIdx.x; n < N_NODES; n += blockDim.x)
    if (gpid[n] == tgt) atomicMin(&best, n);
  __syncthreads();
  int idx = (best == 0x7fffffff) ? 0 : best;   // argmax(all False) == 0
  for (int j = threadIdx.x; j < HH; j += blockDim.x)
    fg[(size_t)b * 256 + HH + j] = node_emb[(size_t)idx * HH + j];
}

// --------------------------------- LSTM ------------------------------------
// 16 blocks x 256 threads; block owns 16 batch rows. Whh (f16) async-staged
// into LDS once; h LDS-resident; c and all four gates live in registers:
// wave w's 4 accumulators are exactly gates i/f/g/o for hidden column
// j = 16w + (lane&15).
__global__ void k_lstm(const float* __restrict__ xW,      // (B*T,512), biases folded
                       const _Float16* __restrict__ WhhH, // (512,128) f16 K-major
                       const int* __restrict__ lengths,
                       float* __restrict__ ts_emb) {      // (B,T,128) masked
  __shared__ _Float16 swhh[512 * HH];   // 128 KB
  __shared__ float    sh[16][HH];       // 8 KB
  __shared__ int      slen[16];
  const int tid  = threadIdx.x;
  const int lane = tid & 31, wave = tid >> 5;
  const int half = lane >> 4, l15 = lane & 15;
  const int b0   = blockIdx.x * 16;

  if (tid < 16) slen[tid] = lengths[b0 + tid];
  for (int e = tid; e < 16 * HH; e += 256) (&sh[0][0])[e] = 0.f;

  // ---- async-stage Whh(f16) to LDS: GLOBAL_LOAD_ASYNC_TO_LDS_B128 ----
  {
    unsigned    lbase = (unsigned)(size_t)(&swhh[0]);
    const char* gbase = (const char*)WhhH;
    for (int it = 0; it < 32; ++it) {
      int chunk = tid + it * 256;                 // 8192 x 16B = 128 KB
      unsigned    loff = lbase + chunk * 16;
      const char* gp   = gbase + (size_t)chunk * 16;
      asm volatile("global_load_async_to_lds_b128 %0, %1, off"
                   :: "v"(loff), "v"(gp) : "memory");
    }
    asm volatile("s_wait_asynccnt 0x0" ::: "memory");
  }
  __syncthreads();

  const int j = wave * 16 + l15;        // this lane's hidden column
  float creg[8];
#pragma unroll
  for (int r = 0; r < 8; ++r) creg[r] = 0.f;

  for (int t = 0; t < TT; ++t) {
    v8f acc[4] = {};
    for (int kk = 0; kk < HH; kk += 32) {
      v16h a;
      const float* ar = &sh[l15][kk + half * 16];
#pragma unroll
      for (int q = 0; q < 8; ++q) a[q]     = (_Float16)ar[q];
#pragma unroll
      for (int q = 0; q < 8; ++q) a[8 + q] = (_Float16)ar[8 + q];
#pragma unroll
      for (int q = 0; q < 4; ++q) {               // gate q, column j
        const int n = q * 128 + wave * 16 + l15;
        v16h b;
        const _Float16* bp = swhh + (size_t)n * HH + kk + half * 16;
#pragma unroll
        for (int jj = 0; jj < 16; ++jj) b[jj] = bp[jj];
        acc[q] = __builtin_amdgcn_wmma_f32_16x16x32_f16(false, a, false, b,
                                                        (short)0, acc[q], false, false);
      }
    }
    __syncthreads();                    // all reads of h_{t-1} done
#pragma unroll
    for (int r = 0; r < 8; ++r) {
      const int m = r + half * 8;
      const size_t xrow = ((size_t)(b0 + m) * TT + t) * 512;
      float gi = acc[0][r] + xW[xrow + j];
      float gf = acc[1][r] + xW[xrow + 128 + j];
      float gg = acc[2][r] + xW[xrow + 256 + j];
      float go = acc[3][r] + xW[xrow + 384 + j];
      float iv = 1.f / (1.f + __expf(-gi));
      float fv = 1.f / (1.f + __expf(-gf));
      float gv = tanhf(gg);
      float ov = 1.f / (1.f + __expf(-go));
      float cv = fv * creg[r] + iv * gv;
      float hv = ov * tanhf(cv);
      creg[r]  = cv;
      sh[m][j] = hv;
      ts_emb[((size_t)(b0 + m) * TT + t) * HH + j] = (t < slen[m]) ? hv : -99.0f;
    }
    __syncthreads();                    // h_t visible for next step
  }
}

// ------------------------------ fc2 + sigmoid ------------------------------
__global__ void k_risk(const float* __restrict__ x, const float* __restrict__ w,
                       const float* __restrict__ b, float* __restrict__ risk, int rows) {
  int r = blockIdx.x * 8 + (threadIdx.x >> 5);
  if (r >= rows) return;
  int lane = threadIdx.x & 31;
  const float4 xv = *(const float4*)(x + (size_t)r * HH + lane * 4);
  const float4 wv = *(const float4*)(w + lane * 4);
  float s = xv.x * wv.x + xv.y * wv.y + xv.z * wv.z + xv.w * wv.w;
#pragma unroll
  for (int o = 16; o > 0; o >>= 1) s += __shfl_down(s, o, 32);
  if (lane == 0) risk[r] = 1.f / (1.f + __expf(-(s + b[0])));
}

// ===========================================================================
extern "C" void kernel_launch(void* const* d_in, const int* in_sizes, int n_in,
                              void* d_out, int out_size, void* d_ws, size_t ws_size,
                              hipStream_t stream) {
  const float* flat_data = (const float*)d_in[0];
  const float* graph_x   = (const float*)d_in[1];
  const float* ts_data   = (const float*)d_in[2];
  const int*   edge_idx  = (const int*)  d_in[3];
  const int*   gpid      = (const int*)  d_in[4];
  const int*   pid       = (const int*)  d_in[5];
  const int*   lengths   = (const int*)  d_in[6];
  const float* flat_W    = (const float*)d_in[7];
  const float* flat_b    = (const float*)d_in[8];
  const float* gcn1_W    = (const float*)d_in[9];
  const float* gcn1_b    = (const float*)d_in[10];
  const float* bn_gamma  = (const float*)d_in[11];
  const float* bn_beta   = (const float*)d_in[12];
  const float* gcn2_W    = (const float*)d_in[13];
  const float* gcn2_b    = (const float*)d_in[14];
  const float* lstm_Wih  = (const float*)d_in[15];
  const float* lstm_Whh  = (const float*)d_in[16];
  const float* lstm_bih  = (const float*)d_in[17];
  const float* lstm_bhh  = (const float*)d_in[18];
  const float* fc1_W     = (const float*)d_in[19];
  const float* fc1_b     = (const float*)d_in[20];
  const float* fc2_W     = (const float*)d_in[21];
  const float* fc2_b     = (const float*)d_in[22];
  const int* e_src = edge_idx;
  const int* e_dst = edge_idx + N_EDGES;

  float* ws = (float*)d_ws;
  size_t off = 0;
  float* dinv   = ws + off; off += 100352;
  float* bufA   = ws + off; off += (size_t)N_NODES * HH;   // 12.8M
  float* bufB   = ws + off; off += (size_t)N_NODES * HH;   // 12.8M
  float* bnsums = ws + off; off += 256;
  float* fg     = ws + off; off += BB * 256;               // [flat|graph] 256x256
  float* base   = ws + off; off += BB * HH;
  float* bsum   = ws + off; off += 512;
  float* ts_emb = ws + off; off += (size_t)BB * TT * HH;   // 4.19M
  _Float16* h16 = (_Float16*)(ws + off); off += 98304;     // f16 weight area
  float* xW     = bufA;   // overlays bufA+bufB tail; both dead before use

  _Float16* Wh1    = h16;            // (128,128)
  _Float16* Wh2    = h16 + 16384;    // (128,128)
  _Float16* flatWh = h16 + 32768;    // (128,64)
  _Float16* WihH   = h16 + 40960;    // (512,64)
  _Float16* fc1aH  = h16 + 73728;    // (128,256)
  _Float16* fc1bH  = h16 + 106496;   // (128,128)
  _Float16* WhhH   = h16 + 122880;   // (512,128)

  float* risk = (float*)d_out;                     // (B,T)
  float* xout = (float*)d_out + (size_t)BB * TT;   // (B,T,H)

  const size_t NH = (size_t)N_NODES * HH;
  dim3 w2(32, 2);

  // ---- weight prep (f16, K-major) ----
  k_prep_B<<<64,  256, 0, stream>>>(gcn1_W,   HH, 0,   0, Wh1,    HH, HH);
  k_prep_B<<<64,  256, 0, stream>>>(gcn2_W,   HH, 0,   0, Wh2,    HH, HH);
  k_prep_B<<<32,  256, 0, stream>>>(flat_W,   64, 1,   0, flatWh, HH, 64);
  k_prep_B<<<128, 256, 0, stream>>>(lstm_Wih, 64, 1,   0, WihH,  512, 64);
  k_prep_B<<<128, 256, 0, stream>>>(fc1_W,   384, 1,   0, fc1aH,  HH, 256);
  k_prep_B<<<64,  256, 0, stream>>>(fc1_W,   384, 1, 256, fc1bH,  HH, HH);
  k_prep_B<<<256, 256, 0, stream>>>(lstm_Whh, HH, 1,   0, WhhH,  512, HH);

  // ---- degrees -> dinv ----
  k_fill<<<(N_NODES + 255) / 256, 256, 0, stream>>>(dinv, 1.0f, N_NODES);
  k_deg_acc<<<(N_EDGES + 255) / 256, 256, 0, stream>>>(e_dst, dinv, N_EDGES);
  k_rsqrt_inplace<<<(N_NODES + 255) / 256, 256, 0, stream>>>(dinv, N_NODES);

  // ---- GCN layer 1 ----
  k_gemm_f16<<<dim3(3125, 2), w2, 0, stream>>>(graph_x, HH, Wh1,
      nullptr, nullptr, 0, 1, 0, bufA, HH, N_NODES, HH, HH);
  k_agg_init<<<(unsigned)((NH + 255) / 256), 256, 0, stream>>>(bufA, dinv, gcn1_b, bufB);
  k_agg_edges<<<(N_EDGES + 7) / 8, 256, 0, stream>>>(bufA, dinv, e_src, e_dst, bufB, N_EDGES);
  k_fill<<<1, 256, 0, stream>>>(bnsums, 0.0f, 256);
  k_bn_stats<<<(N_NODES + 255) / 256, 128, 0, stream>>>(bufB, bnsums);
  k_bn_relu<<<(unsigned)((NH + 255) / 256), 256, 0, stream>>>(bufB, bnsums, bn_gamma, bn_beta, bufA);

  // ---- GCN layer 2 (node_emb -> bufA) ----
  k_gemm_f16<<<dim3(3125, 2), w2, 0, stream>>>(bufA, HH, Wh2,
      nullptr, nullptr, 0, 1, 0, bufB, HH, N_NODES, HH, HH);
  k_agg_init<<<(unsigned)((NH + 255) / 256), 256, 0, stream>>>(bufB, dinv, gcn2_b, bufA);
  k_agg_edges<<<(N_EDGES + 7) / 8, 256, 0, stream>>>(bufB, dinv, e_src, e_dst, bufA, N_EDGES);

  // ---- flat encoder -> fg[:, :128]; graph join -> fg[:, 128:] ----
  k_gemm_f16<<<dim3(8, 2), w2, 0, stream>>>(flat_data, 64, flatWh,
      flat_b, nullptr, 0, 1, 0, fg, 256, BB, HH, 64);
  k_join<<<BB, 256, 0, stream>>>(gpid, pid, bufA, fg);   // consumes node_emb

  // ---- LSTM input projection (biases folded); xW overlays bufA/bufB ----
  k_add_vec<<<2, 256, 0, stream>>>(bsum, lstm_bih, lstm_bhh, 512);
  k_gemm_f16<<<dim3(1024, 8), w2, 0, stream>>>(ts_data, 64, WihH,
      bsum, nullptr, 0, 1, 0, xW, 512, BB * TT, 512, 64);

  // ---- t-invariant half of fc1: base = fg @ fc1_W[:, :256]^T + fc1_b ----
  k_gemm_f16<<<dim3(8, 2), w2, 0, stream>>>(fg, 256, fc1aH,
      fc1_b, nullptr, 0, 1, 0, base, HH, BB, HH, 256);

  // ---- recurrent LSTM (async LDS-staged Whh, register gates) ----
  k_lstm<<<16, 256, 0, stream>>>(xW, WhhH, lengths, ts_emb);

  // ---- x = relu(base[b] + ts_emb @ fc1_W[:, 256:]^T), rowShift = log2(T) ----
  k_gemm_f16<<<dim3(1024, 2), w2, 0, stream>>>(ts_emb, HH, fc1bH,
      nullptr, base, 7, HH, 1, xout, HH, BB * TT, HH, HH);

  // ---- risk = sigmoid(x @ fc2_W^T + fc2_b) ----
  k_risk<<<(BB * TT) / 8, 256, 0, stream>>>(xout, fc2_W, fc2_b, risk, BB * TT);
}
